// MultiHeadAttentionCosSquareformerNew_51745765982279
// MI455X (gfx1250) — matrix operand order repaired and checked by reading, not compiled, not run on hardware
//
#include <hip/hip_runtime.h>

// ---------------------------------------------------------------------------
// CDNA5 / gfx1250 cosformer-style linear attention.
// - All GEMMs: v_wmma_f32_16x16x32_bf16 (fp32 accumulate)
// - Streaming K^T V contraction: double-buffered LDS slabs filled with
//   global_load_async_to_lds_b128, consumed with ds_load_b128.
// - All intermediate layouts chosen so WMMA operands are contiguous per lane.
// ---------------------------------------------------------------------------

typedef __bf16 bf16;
typedef __attribute__((ext_vector_type(16))) __bf16 v16bf;
typedef __attribute__((ext_vector_type(8)))  __bf16 v8bf;
typedef __attribute__((ext_vector_type(8)))  float  v8f;

#define PI_F 3.1415f
// B=4, S=4096, D=64, H=16, D*H=1024, B*S=16384, B*H=64

__device__ __forceinline__ float elu1(float x) {
  return x > 0.f ? x + 1.f : __expf(x);   // elu(x)+1
}

__device__ __forceinline__ v8f wmma_bf16(v16bf a, v16bf b, v8f c) {
  return __builtin_amdgcn_wmma_f32_16x16x32_bf16(
      false, a, false, b, (short)0, c, false, false);
}

__device__ __forceinline__ v16bf cat16(v8bf l, v8bf h) {
  return __builtin_shufflevector(l, h, 0,1,2,3,4,5,6,7,8,9,10,11,12,13,14,15);
}

// Low 32 bits of a flat pointer to LDS == LDS byte offset (aperture layout).
__device__ __forceinline__ unsigned lds_off(const void* p) {
  return (unsigned)(uintptr_t)p;
}

// GLOBAL_LOAD_ASYNC_TO_LDS_B128 (ISA 15.18.3 op 98): per-lane 16B gmem->LDS,
// tracked by ASYNCcnt.
__device__ __forceinline__ void async_ld_b128(unsigned lds_byte_off, const void* gaddr) {
  asm volatile("global_load_async_to_lds_b128 %0, %1, off"
               :: "v"(lds_byte_off), "v"(gaddr) : "memory");
}
__device__ __forceinline__ void wait_async0() {
  asm volatile("s_wait_asynccnt 0" ::: "memory");
}

// ---------------------------------------------------------------------------
// Kernel 0: transpose + bf16-convert a weight matrix.
// WT[c*R + d] = (bf16) W[d*C + c]   (out has R*C elements, coalesced writes)
// ---------------------------------------------------------------------------
__global__ void wtrans_kernel(const float* __restrict__ W, bf16* __restrict__ WT,
                              int R, int C) {
  const int idx = blockIdx.x * blockDim.x + threadIdx.x;
  if (idx >= R * C) return;
  const int d = idx % R;           // inner (contraction) index of output row
  const int c = idx / R;
  WT[idx] = (bf16)W[(size_t)d * C + c];
}

// ---------------------------------------------------------------------------
// Kernel 1: Y = X(16384x64) @ W(64x1024) + b, epilogue per projection.
// which: 0=Q (elu+1 -> Q,QCS token-major), 1=K (elu+1 -> Kt,KSt ch-major),
//        2=V (-> Vt ch-major).  WT is the pre-transposed bf16 weight (1024x64).
// One wave per 16x16 tile; A = two float4-pairs, B = two v8bf loads.
// ---------------------------------------------------------------------------
__global__ void qkv_proj_kernel(const float* __restrict__ X,
                                const bf16* __restrict__ WT,
                                const float* __restrict__ bias,
                                bf16* __restrict__ out0,
                                bf16* __restrict__ out1,
                                int which) {
  const int wave = threadIdx.x >> 5;
  const int lane = threadIdx.x & 31;
  const int lo = lane & 15, hi = lane >> 4;
  const int tile = blockIdx.x * (blockDim.x >> 5) + wave;
  if (tile >= 65536) return;                 // wave-uniform
  const int t0 = (tile >> 6) * 16;           // token tile
  const int c0 = (tile & 63) * 16;           // channel tile

  const float* xrow = X + (size_t)(t0 + lo) * 64;
  v8f acc = {};
#pragma unroll
  for (int k0 = 0; k0 < 64; k0 += 32) {
    const float4 ca0 = *(const float4*)(xrow + k0 + hi * 8);
    const float4 ca1 = *(const float4*)(xrow + k0 + hi * 8 + 4);
    const float4 cb0 = *(const float4*)(xrow + k0 + 16 + hi * 8);
    const float4 cb1 = *(const float4*)(xrow + k0 + 16 + hi * 8 + 4);
    v16bf a;
    a[0]=(bf16)ca0.x; a[1]=(bf16)ca0.y; a[2]=(bf16)ca0.z; a[3]=(bf16)ca0.w;
    a[4]=(bf16)ca1.x; a[5]=(bf16)ca1.y; a[6]=(bf16)ca1.z; a[7]=(bf16)ca1.w;
    a[8]=(bf16)cb0.x; a[9]=(bf16)cb0.y; a[10]=(bf16)cb0.z; a[11]=(bf16)cb0.w;
    a[12]=(bf16)cb1.x; a[13]=(bf16)cb1.y; a[14]=(bf16)cb1.z; a[15]=(bf16)cb1.w;
    const v16bf bm = cat16(*(const v8bf*)(WT + (size_t)(c0 + lo) * 64 + k0 + hi * 8),
                           *(const v8bf*)(WT + (size_t)(c0 + lo) * 64 + k0 + 16 + hi * 8));
    acc = wmma_bf16(a, bm, acc);
  }

  const float bc = bias[c0 + lo];
  if (which == 2) {                                    // V -> Vt[(b*16+h)*64+z][s]
    const int c = c0 + lo, b = t0 >> 12, s0 = (t0 & 4095) + hi * 8;
    v8bf st;
#pragma unroll
    for (int v = 0; v < 8; ++v) st[v] = (bf16)(acc[v] + bc);
    *(v8bf*)(out0 + ((size_t)b * 1024 + c) * 4096 + s0) = st;
  } else if (which == 1) {                             // K -> Kt, KSt (ch-major)
    const int c = c0 + lo, b = t0 >> 12, s0 = (t0 & 4095) + hi * 8;
    v8bf st0, st1;
#pragma unroll
    for (int v = 0; v < 8; ++v) {
      const float p = PI_F * (float)(s0 + v) / 4096.f;
      const float y = elu1(acc[v] + bc);
      st0[v] = (bf16)y;
      st1[v] = (bf16)(y * __sinf(p));
    }
    const size_t o = ((size_t)b * 1024 + c) * 4096 + s0;
    *(v8bf*)(out0 + o) = st0;
    *(v8bf*)(out1 + o) = st1;
  } else {                                             // Q -> Q, QCS token-major
#pragma unroll
    for (int v = 0; v < 8; ++v) {
      const int token = t0 + hi * 8 + v;
      const float p = PI_F * (float)(token & 4095) / 4096.f;
      const float y = elu1(acc[v] + bc);
      const size_t o = (size_t)token * 1024 + c0 + lo;
      out0[o] = (bf16)y;
      out1[o] = (bf16)(y * (__cosf(p) + __sinf(p)));
    }
  }
}

// ---------------------------------------------------------------------------
// Kernel 2: per (b,h): KV = K^T V, KVS = KS^T V (64x64), contraction S=4096.
// One block (8 waves) per bh. Double-buffered LDS slabs (32 s x 64 ch bf16)
// filled with global_load_async_to_lds_b128; each slab read once from HBM/L2
// and consumed by all 8 waves via ds_load_b128. Outputs stored z-major so the
// D fragment is one contiguous v8bf store.
// ---------------------------------------------------------------------------
__global__ void kv_kernel(const bf16* __restrict__ Kt, const bf16* __restrict__ KSt,
                          const bf16* __restrict__ Vt,
                          bf16* __restrict__ KVt, bf16* __restrict__ KVSt) {
  __shared__ bf16 lds_k[2][64 * 32];
  __shared__ bf16 lds_s[2][64 * 32];
  __shared__ bf16 lds_v[2][64 * 32];

  const int bh = blockIdx.x;                 // 0..63
  const int tid = threadIdx.x;               // 256 threads
  const int wave = tid >> 5;
  const int lane = tid & 31;
  const int lo = lane & 15, hi = lane >> 4;

  // Wave tile assignment: x-strip + two z tiles.
  const int x0 = (wave >> 1) * 16;
  const int z0a = (wave & 1) * 32;
  const int z0b = z0a + 16;

  // Loader assignment: each thread moves one 16B chunk per array per slab.
  const int row = tid >> 2;                  // 0..63 (channel within head)
  const int qof = (tid & 3) * 8;             // 0,8,16,24 (s offset)
  const size_t gbase = ((size_t)bh * 64 + row) * 4096 + qof;

  auto issue = [&](int chunk, int bsel) {
    const size_t g = gbase + (size_t)chunk * 32;
    const unsigned l = (unsigned)((row * 32 + qof) * (int)sizeof(bf16));
    async_ld_b128(lds_off((const char*)lds_k[bsel] + l), Kt + g);
    async_ld_b128(lds_off((const char*)lds_s[bsel] + l), KSt + g);
    async_ld_b128(lds_off((const char*)lds_v[bsel] + l), Vt + g);
  };

  v8f accKa = {}, accKb = {}, accSa = {}, accSb = {};
  issue(0, 0);
  for (int c = 0; c < 128; ++c) {
    wait_async0();                           // own slab-c transfers done
    __syncthreads();                         // slab c visible; buf c^1 free
    if (c + 1 < 128) issue(c + 1, (c + 1) & 1);

    const int bsel = c & 1;
    const bf16* lk = lds_k[bsel];
    const bf16* ls = lds_s[bsel];
    const bf16* lv = lds_v[bsel];
    const v16bf aK = cat16(*(const v8bf*)(lk + (x0 + lo) * 32 + hi * 8),
                           *(const v8bf*)(lk + (x0 + lo) * 32 + 16 + hi * 8));
    const v16bf aS = cat16(*(const v8bf*)(ls + (x0 + lo) * 32 + hi * 8),
                           *(const v8bf*)(ls + (x0 + lo) * 32 + 16 + hi * 8));
    const v16bf bva = cat16(*(const v8bf*)(lv + (z0a + lo) * 32 + hi * 8),
                            *(const v8bf*)(lv + (z0a + lo) * 32 + 16 + hi * 8));
    const v16bf bvb = cat16(*(const v8bf*)(lv + (z0b + lo) * 32 + hi * 8),
                            *(const v8bf*)(lv + (z0b + lo) * 32 + 16 + hi * 8));
    accKa = wmma_bf16(aK, bva, accKa);
    accKb = wmma_bf16(aK, bvb, accKb);
    accSa = wmma_bf16(aS, bva, accSa);
    accSb = wmma_bf16(aS, bvb, accSb);
  }

  v8bf sKa, sKb, sSa, sSb;
#pragma unroll
  for (int v = 0; v < 8; ++v) {
    sKa[v] = (bf16)accKa[v]; sKb[v] = (bf16)accKb[v];
    sSa[v] = (bf16)accSa[v]; sSb[v] = (bf16)accSb[v];
  }
  const size_t oa = ((size_t)bh * 64 + z0a + lo) * 64 + x0 + hi * 8;
  const size_t ob = ((size_t)bh * 64 + z0b + lo) * 64 + x0 + hi * 8;
  *(v8bf*)(KVt + oa)  = sKa;  *(v8bf*)(KVt + ob)  = sKb;
  *(v8bf*)(KVSt + oa) = sSa;  *(v8bf*)(KVSt + ob) = sSb;
}

// ---------------------------------------------------------------------------
// Kernel 2b: ksum[bh*64+x] = sum_s Kt row ; ksums likewise (contiguous rows).
// ---------------------------------------------------------------------------
__global__ void ksum_kernel(const bf16* __restrict__ Kt, const bf16* __restrict__ KSt,
                            float* __restrict__ ksum, float* __restrict__ ksums) {
  const int idx = blockIdx.x * blockDim.x + threadIdx.x;   // 4096
  if (idx >= 4096) return;
  const bf16* r0 = Kt + (size_t)idx * 4096;
  const bf16* r1 = KSt + (size_t)idx * 4096;
  float s = 0.f, s2 = 0.f;
  for (int i = 0; i < 512; ++i) {
    __builtin_prefetch(r0 + i * 8 + 512, 0, 1);
    const v8bf a = *(const v8bf*)(r0 + i * 8);
    const v8bf b = *(const v8bf*)(r1 + i * 8);
#pragma unroll
    for (int j = 0; j < 8; ++j) { s += (float)a[j]; s2 += (float)b[j]; }
  }
  ksum[idx] = s;
  ksums[idx] = s2;
}

// ---------------------------------------------------------------------------
// Kernel 2c: denomInv[t*16+h] = 1 / (q.ksum + qcs.ksums + 1e-5)
// ---------------------------------------------------------------------------
__global__ void denom_kernel(const bf16* __restrict__ Q, const bf16* __restrict__ QCS,
                             const float* __restrict__ ksum, const float* __restrict__ ksums,
                             float* __restrict__ denomInv) {
  const int idx = blockIdx.x * blockDim.x + threadIdx.x;   // 262144
  if (idx >= 262144) return;
  const int t = idx >> 4, h = idx & 15;
  const int b = t >> 12;
  const bf16* qrow  = Q   + (size_t)t * 1024 + h * 64;
  const bf16* qcrow = QCS + (size_t)t * 1024 + h * 64;
  const float* ks  = ksum  + (size_t)(b * 16 + h) * 64;
  const float* kss = ksums + (size_t)(b * 16 + h) * 64;
  float acc = 1e-5f;
#pragma unroll 8
  for (int x = 0; x < 64; ++x)
    acc += (float)qrow[x] * ks[x] + (float)qcrow[x] * kss[x];
  denomInv[idx] = 1.0f / acc;
}

// ---------------------------------------------------------------------------
// Kernel 3: O = (Q @ KV + QCS @ KVS) * denomInv  (per b,h; contraction = 64)
// A: two v8bf loads from token-major Q/QCS; B: two v8bf loads from z-major kv.
// ---------------------------------------------------------------------------
__global__ void attn_apply_kernel(const bf16* __restrict__ Q, const bf16* __restrict__ QCS,
                                  const bf16* __restrict__ KVt, const bf16* __restrict__ KVSt,
                                  const float* __restrict__ denomInv, bf16* __restrict__ O) {
  const int wave = threadIdx.x >> 5;
  const int lane = threadIdx.x & 31;
  const int lo = lane & 15, hi = lane >> 4;
  const int tile = blockIdx.x * (blockDim.x >> 5) + wave;
  if (tile >= 65536) return;
  const int bh = tile >> 10;
  const int rest = tile & 1023;
  const int tt = rest >> 2, zt = rest & 3;
  const int b = bh >> 4, h = bh & 15;
  const int t0 = b * 4096 + tt * 16;
  const int z0 = zt * 16;

  const bf16* qrow  = Q   + (size_t)(t0 + lo) * 1024 + h * 64;
  const bf16* qcrow = QCS + (size_t)(t0 + lo) * 1024 + h * 64;
  v8f acc = {};
#pragma unroll
  for (int k0 = 0; k0 < 64; k0 += 32) {
    const v16bf a  = cat16(*(const v8bf*)(qrow + k0 + hi * 8),
                           *(const v8bf*)(qrow + k0 + 16 + hi * 8));
    const v16bf a2 = cat16(*(const v8bf*)(qcrow + k0 + hi * 8),
                           *(const v8bf*)(qcrow + k0 + 16 + hi * 8));
    const v16bf bm  = cat16(*(const v8bf*)(KVt  + ((size_t)bh * 64 + z0 + lo) * 64 + k0 + hi * 8),
                            *(const v8bf*)(KVt  + ((size_t)bh * 64 + z0 + lo) * 64 + k0 + 16 + hi * 8));
    const v16bf bm2 = cat16(*(const v8bf*)(KVSt + ((size_t)bh * 64 + z0 + lo) * 64 + k0 + hi * 8),
                            *(const v8bf*)(KVSt + ((size_t)bh * 64 + z0 + lo) * 64 + k0 + 16 + hi * 8));
    acc = wmma_bf16(a,  bm,  acc);
    acc = wmma_bf16(a2, bm2, acc);
  }
#pragma unroll
  for (int v = 0; v < 8; ++v) {
    const int token = t0 + hi * 8 + v;
    const float dv = denomInv[(size_t)token * 16 + h];
    O[(size_t)token * 1024 + h * 64 + z0 + lo] = (bf16)(acc[v] * dv);
  }
}

// ---------------------------------------------------------------------------
// Kernel 4: out = O(16384x1024) @ dense_w(1024x64) + dense_b (fp32 out).
// DT is pre-transposed bf16 dense weight (64 x 1024, n-major).
// ---------------------------------------------------------------------------
__global__ void dense_kernel(const bf16* __restrict__ O, const bf16* __restrict__ DT,
                             const float* __restrict__ bias, float* __restrict__ out) {
  const int wave = threadIdx.x >> 5;
  const int lane = threadIdx.x & 31;
  const int lo = lane & 15, hi = lane >> 4;
  const int tile = blockIdx.x * (blockDim.x >> 5) + wave;
  if (tile >= 4096) return;
  const int t0 = (tile >> 2) * 16;
  const int n0 = (tile & 3) * 16;

  const bf16* orow = O + (size_t)(t0 + lo) * 1024;
  const bf16* wrow = DT + (size_t)(n0 + lo) * 1024;
  v8f acc = {};
  for (int k0 = 0; k0 < 1024; k0 += 32) {
    const v16bf a = cat16(*(const v8bf*)(orow + k0 + hi * 8),
                          *(const v8bf*)(orow + k0 + 16 + hi * 8));
    const v16bf bm = cat16(*(const v8bf*)(wrow + k0 + hi * 8),
                           *(const v8bf*)(wrow + k0 + 16 + hi * 8));
    acc = wmma_bf16(a, bm, acc);
  }
#pragma unroll
  for (int v = 0; v < 8; ++v)
    out[(size_t)(t0 + hi * 8 + v) * 64 + n0 + lo] = acc[v] + bias[n0 + lo];
}

// ---------------------------------------------------------------------------
extern "C" void kernel_launch(void* const* d_in, const int* in_sizes, int n_in,
                              void* d_out, int out_size, void* d_ws, size_t ws_size,
                              hipStream_t stream) {
  (void)in_sizes; (void)n_in; (void)out_size; (void)ws_size;
  const float* query   = (const float*)d_in[0];
  const float* key     = (const float*)d_in[1];
  const float* value   = (const float*)d_in[2];
  /* d_in[3] attn_mask unused */
  const float* wq_w    = (const float*)d_in[4];
  const float* wq_b    = (const float*)d_in[5];
  const float* wk_w    = (const float*)d_in[6];
  const float* wk_b    = (const float*)d_in[7];
  const float* wv_w    = (const float*)d_in[8];
  const float* wv_b    = (const float*)d_in[9];
  const float* dense_w = (const float*)d_in[10];
  const float* dense_b = (const float*)d_in[11];

  char* ws = (char*)d_ws;
  const size_t SZ = (size_t)16384 * 1024 * 2;   // 32 MiB per bf16 matrix
  bf16* Q    = (bf16*)(ws + 0 * SZ);            // token-major
  bf16* QCS  = (bf16*)(ws + 1 * SZ);            // token-major
  bf16* Kt   = (bf16*)(ws + 2 * SZ);            // (b,h,x,s)
  bf16* KSt  = (bf16*)(ws + 3 * SZ);            // (b,h,x,s)
  bf16* Vt   = (bf16*)(ws + 4 * SZ);            // (b,h,z,s)
  bf16* O    = (bf16*)(ws + 5 * SZ);            // token-major
  char* p = ws + 6 * SZ;
  bf16* KVt   = (bf16*)(p);              p += 524288;   // (bh, z, x)
  bf16* KVSt  = (bf16*)(p);              p += 524288;
  float* ksum  = (float*)(p);            p += 16384;
  float* ksums = (float*)(p);            p += 16384;
  float* denomInv = (float*)(p);         p += 1048576;
  bf16* WqT = (bf16*)(p);                p += 131072;   // 1024 x 64
  bf16* WkT = (bf16*)(p);                p += 131072;
  bf16* WvT = (bf16*)(p);                p += 131072;
  bf16* DT  = (bf16*)(p);                p += 131072;   // 64 x 1024
  float* out = (float*)d_out;

  const dim3 blk(256);   // 8 waves (wave32)
  wtrans_kernel<<<dim3(256), blk, 0, stream>>>(wq_w, WqT, 64, 1024);
  wtrans_kernel<<<dim3(256), blk, 0, stream>>>(wk_w, WkT, 64, 1024);
  wtrans_kernel<<<dim3(256), blk, 0, stream>>>(wv_w, WvT, 64, 1024);
  wtrans_kernel<<<dim3(256), blk, 0, stream>>>(dense_w, DT, 1024, 64);

  qkv_proj_kernel<<<dim3(8192), blk, 0, stream>>>(query, WqT, wq_b, Q,  QCS, 0);
  qkv_proj_kernel<<<dim3(8192), blk, 0, stream>>>(key,   WkT, wk_b, Kt, KSt, 1);
  qkv_proj_kernel<<<dim3(8192), blk, 0, stream>>>(value, WvT, wv_b, Vt, Vt,  2);

  kv_kernel<<<dim3(64), blk, 0, stream>>>(Kt, KSt, Vt, KVt, KVSt);
  ksum_kernel<<<dim3(16), blk, 0, stream>>>(Kt, KSt, ksum, ksums);
  denom_kernel<<<dim3(1024), blk, 0, stream>>>(Q, QCS, ksum, ksums, denomInv);

  attn_apply_kernel<<<dim3(8192), blk, 0, stream>>>(Q, QCS, KVt, KVSt, denomInv, O);
  dense_kernel<<<dim3(512), blk, 0, stream>>>(O, DT, dense_b, out);
}
